// MyModel_67834713473342
// MI455X (gfx1250) — matrix operand order repaired and checked by reading, not compiled, not run on hardware
//
#include <hip/hip_runtime.h>
#include <hip/hip_bf16.h>
#include <math.h>

typedef __attribute__((ext_vector_type(16))) _Float16 v16h;
typedef __attribute__((ext_vector_type(8)))  _Float16 v8h;
typedef __attribute__((ext_vector_type(8)))  float    v8f;

#define NC_ 65536
#define EC_ 262144
#define NM_ 131072
#define EM_ 524288
#define G_  256
#define B_  32
#define D_  512
#define KE_ 300
#define KP_ 320
#define NPG_ 512          // nodes per comment graph (NM/G)
#define KKEEP_ 103        // ceil(0.2 * 512)
#define NPB_ 2048         // content nodes per batch (NC/B)
#define QSCALE_ 0.08838834764831843f  // 128^-0.5

#define CDIV(a,b) (((a)+(b)-1)/(b))

// ---------------- utility kernels ----------------

__global__ void zero_kernel(float* p, int n) {
  int i = blockIdx.x * blockDim.x + threadIdx.x;
  if (i < n) p[i] = 0.f;
}

// weights f32 [N,K] -> f16 [N,Kp] (zero padded)
__global__ void wconv_kernel(const float* __restrict__ w, _Float16* __restrict__ out,
                             int N, int K, int Kp) {
  int i = blockIdx.x * blockDim.x + threadIdx.x;
  if (i >= N * Kp) return;
  int n = i / Kp, k = i % Kp;
  out[i] = (k < K) ? (_Float16)w[(size_t)n * K + k] : (_Float16)0.f;
}

// gather emb rows -> f16 padded
__global__ void gather_half_kernel(const float* __restrict__ emb, const int* __restrict__ idx,
                                   _Float16* __restrict__ out, long long total, int Kp) {
  long long i = (long long)blockIdx.x * blockDim.x + threadIdx.x;
  if (i >= total) return;
  int node = (int)(i / Kp), k = (int)(i % Kp);
  out[i] = (k < KE_) ? (_Float16)emb[(size_t)idx[node] * KE_ + k] : (_Float16)0.f;
}

__global__ void relu_half_kernel(const float* __restrict__ in, _Float16* __restrict__ out,
                                 long long n) {
  long long i = (long long)blockIdx.x * blockDim.x + threadIdx.x;
  if (i < n) out[i] = (_Float16)fmaxf(in[i], 0.f);
}

// ---------------- WMMA GEMM:  H[M,512] = X[M,Kp] @ W[512,Kp]^T ----------------
__global__ __launch_bounds__(128)
void gemm_f16_wmma(const _Float16* __restrict__ X, const _Float16* __restrict__ W,
                   float* __restrict__ Hout, int Kpad) {
  const int lane = threadIdx.x & 31;
  const int wave = threadIdx.x >> 5;
  const int hl   = lane >> 4;      // half-wave: 0 or 1
  const int l16  = lane & 15;
  const size_t arow = (size_t)blockIdx.x * 64 + wave * 16 + l16;
  const int ncol = blockIdx.y * 64;
  v8f acc0 = {}, acc1 = {}, acc2 = {}, acc3 = {};
  const _Float16* ap  = X + arow * Kpad + hl * 16;
  const _Float16* bp0 = W + (size_t)(ncol +  0 + l16) * Kpad + hl * 16;
  const _Float16* bp1 = W + (size_t)(ncol + 16 + l16) * Kpad + hl * 16;
  const _Float16* bp2 = W + (size_t)(ncol + 32 + l16) * Kpad + hl * 16;
  const _Float16* bp3 = W + (size_t)(ncol + 48 + l16) * Kpad + hl * 16;
  for (int k0 = 0; k0 < Kpad; k0 += 32) {
    union { v16h v; v8h h[2]; } a;
    a.h[0] = *(const v8h*)(ap);        // K = k0 + hl*16 + [0,8)
    a.h[1] = *(const v8h*)(ap + 8);    // K = k0 + hl*16 + 8 + [0,8)
    v16h b0 = *(const v16h*)(bp0);     // K = k0 + hl*16 + [0,16), N = ncol+j*16+l16
    v16h b1 = *(const v16h*)(bp1);
    v16h b2 = *(const v16h*)(bp2);
    v16h b3 = *(const v16h*)(bp3);
    acc0 = __builtin_amdgcn_wmma_f32_16x16x32_f16(false, a.v, false, b0, (short)0, acc0, false, false);
    acc1 = __builtin_amdgcn_wmma_f32_16x16x32_f16(false, a.v, false, b1, (short)0, acc1, false, false);
    acc2 = __builtin_amdgcn_wmma_f32_16x16x32_f16(false, a.v, false, b2, (short)0, acc2, false, false);
    acc3 = __builtin_amdgcn_wmma_f32_16x16x32_f16(false, a.v, false, b3, (short)0, acc3, false, false);
    ap += 32; bp0 += 32; bp1 += 32; bp2 += 32; bp3 += 32;
  }
  const size_t rbase = (size_t)blockIdx.x * 64 + wave * 16 + 8 * hl;
#pragma unroll
  for (int r = 0; r < 8; ++r) {
    size_t row = rbase + r;
    Hout[row * D_ + ncol +  0 + l16] = acc0[r];
    Hout[row * D_ + ncol + 16 + l16] = acc1[r];
    Hout[row * D_ + ncol + 32 + l16] = acc2[r];
    Hout[row * D_ + ncol + 48 + l16] = acc3[r];
  }
}

// ---------------- GCN helpers ----------------

__global__ void edge_deg_unit_kernel(const int* __restrict__ edge, float* __restrict__ deg, int E) {
  int e = blockIdx.x * blockDim.x + threadIdx.x;
  if (e >= E) return;
  atomicAdd(&deg[edge[E + e]], 1.f);
}

__global__ void edge_deg_mask_kernel(const int* __restrict__ edge, const int* __restrict__ keep,
                                     float* __restrict__ deg, int E) {
  int e = blockIdx.x * blockDim.x + threadIdx.x;
  if (e >= E) return;
  int s = edge[e], d = edge[E + e];
  if (keep[s] && keep[d]) atomicAdd(&deg[d], 1.f);
}

__global__ void degfin_kernel(float* __restrict__ deg, float* __restrict__ dis, int n) {
  int i = blockIdx.x * blockDim.x + threadIdx.x;
  if (i >= n) return;
  float d = deg[i] + 1.f;
  deg[i] = d;
  dis[i] = rsqrtf(d);
}

// out[i,c] = H[i,c]/deg[i] + b[c]  (self-loop term + bias)
__global__ void init_gcn_kernel(float* __restrict__ out, const float* __restrict__ H,
                                const float* __restrict__ deg, const float* __restrict__ b,
                                long long total) {
  long long i = (long long)blockIdx.x * blockDim.x + threadIdx.x;
  if (i >= total) return;
  int node = (int)(i >> 9), c = (int)(i & 511);
  out[i] = H[i] / deg[node] + b[c];
}

__global__ void edge_agg_unit_kernel(const int* __restrict__ edge, const float* __restrict__ H,
                                     const float* __restrict__ dis, float* __restrict__ out, int E) {
  int e = blockIdx.x;
  int s = edge[e], d = edge[E + e];
  float norm = dis[s] * dis[d];
  const float* hs = H + (size_t)s * D_;
  float* od = out + (size_t)d * D_;
  for (int c = threadIdx.x; c < D_; c += blockDim.x)
    atomicAdd(&od[c], hs[c] * norm);
}

__global__ void edge_agg_mask_kernel(const int* __restrict__ edge, const int* __restrict__ keep,
                                     const float* __restrict__ H, const float* __restrict__ dis,
                                     float* __restrict__ out, int E) {
  int e = blockIdx.x;
  int s = edge[e], d = edge[E + e];
  if (!(keep[s] && keep[d])) return;
  float norm = dis[s] * dis[d];
  const float* hs = H + (size_t)s * D_;
  float* od = out + (size_t)d * D_;
  for (int c = threadIdx.x; c < D_; c += blockDim.x)
    atomicAdd(&od[c], hs[c] * norm);
}

// mean pool content graphs (nodes contiguous per batch) with relu
__global__ void meanpool_kernel(const float* __restrict__ agg, float* __restrict__ news) {
  int b = blockIdx.x;
  int c = blockIdx.y * 64 + threadIdx.x;
  float acc = 0.f;
  const float* base = agg + (size_t)b * NPB_ * D_ + c;
  for (int j = 0; j < NPB_; ++j) acc += fmaxf(base[(size_t)j * D_], 0.f);
  news[b * D_ + c] = acc / (float)NPB_;
}

// ---------------- score path (linearized GCN->scalar) ----------------

// wsv[k<300] = sum_n wtr[n,k]*w_score[n] ; wsv[300] = btr.w_score + b_score
__global__ void scorevec_kernel(const float* __restrict__ wtr, const float* __restrict__ btr,
                                const float* __restrict__ w_score, const float* __restrict__ b_score,
                                float* __restrict__ wsv) {
  int k = blockIdx.x * blockDim.x + threadIdx.x;
  if (k < KE_) {
    float acc = 0.f;
    for (int n = 0; n < D_; ++n) acc += wtr[(size_t)n * KE_ + k] * w_score[n];
    wsv[k] = acc;
  } else if (k == KE_) {
    float acc = b_score[0];
    for (int n = 0; n < D_; ++n) acc += btr[n] * w_score[n];
    wsv[KE_] = acc;
  }
}

__global__ void tvec_kernel(const float* __restrict__ emb, const int* __restrict__ xidx,
                            const float* __restrict__ wsv, float* __restrict__ t) {
  int i = blockIdx.x * blockDim.x + threadIdx.x;
  if (i >= NM_) return;
  const float* row = emb + (size_t)xidx[i] * KE_;
  float acc = 0.f;
  for (int k = 0; k < KE_; ++k) acc += row[k] * wsv[k];
  t[i] = acc;
}

__global__ void score_init_kernel(const float* __restrict__ t, const float* __restrict__ deg,
                                  const float* __restrict__ wsv, float* __restrict__ score) {
  int i = blockIdx.x * blockDim.x + threadIdx.x;
  if (i >= NM_) return;
  score[i] = t[i] / deg[i] + wsv[KE_];
}

__global__ void edge_score_kernel(const int* __restrict__ edge, const float* __restrict__ t,
                                  const float* __restrict__ dis, float* __restrict__ score, int E) {
  int e = blockIdx.x * blockDim.x + threadIdx.x;
  if (e >= E) return;
  int s = edge[e], d = edge[E + e];
  atomicAdd(&score[d], t[s] * dis[s] * dis[d]);
}

// exact top-k per graph: rank = #{score_j > score_i} + #{score_j == score_i, j < i}
__global__ void keep_kernel(const float* __restrict__ score, int* __restrict__ keep) {
  __shared__ float sc[NPG_];
  int g = blockIdx.x, t = threadIdx.x;
  int i = g * NPG_ + t;
  sc[t] = score[i];
  __syncthreads();
  float my = sc[t];
  int rank = 0;
  for (int j = 0; j < NPG_; ++j) {
    float o = sc[j];
    rank += (o > my) || (o == my && j < t);
  }
  keep[i] = (rank < KKEEP_) ? 1 : 0;
}

// ---------------- dense linear (small mats) ----------------
__global__ void linear_kernel(const float* __restrict__ X, const int* __restrict__ rowmap,
                              const float* __restrict__ W, const float* __restrict__ bias,
                              float* __restrict__ out, int rows, int N, int K,
                              float scale, int dorelu) {
  int idx = blockIdx.x * blockDim.x + threadIdx.x;
  if (idx >= rows * N) return;
  int r = idx / N, n = idx % N;
  int rr = rowmap ? rowmap[r] : r;
  const float* x = X + (size_t)rr * K;
  const float* w = W + (size_t)n * K;
  float acc = bias ? bias[n] : 0.f;
  for (int k = 0; k < K; ++k) acc += x[k] * w[k];
  acc *= scale;
  if (dorelu) acc = fmaxf(acc, 0.f);
  out[(size_t)r * N + n] = acc;
}

// ---------------- node_attn (algebraically reduced) ----------------

// u[g,h,c] = sum_d wk[h*128+d, c] * qh[g, h*128+d]
__global__ void attn_u_kernel(const float* __restrict__ qh, const float* __restrict__ wk,
                              float* __restrict__ u) {
  int idx = blockIdx.x * blockDim.x + threadIdx.x; // G*4*512
  if (idx >= G_ * 4 * D_) return;
  int c = idx & 511, gh = idx >> 9, h = gh & 3, g = gh >> 2;
  const float* q = qh + (size_t)g * D_ + h * 128;
  float acc = 0.f;
  for (int d = 0; d < 128; ++d) acc += wk[(size_t)(h * 128 + d) * D_ + c] * q[d];
  u[idx] = acc;
}

__global__ void attn_c0_kernel(const float* __restrict__ qh, const float* __restrict__ bk,
                               float* __restrict__ c0) {
  int idx = blockIdx.x * blockDim.x + threadIdx.x; // G*4
  if (idx >= G_ * 4) return;
  int h = idx & 3, g = idx >> 2;
  const float* q = qh + (size_t)g * D_ + h * 128;
  float acc = 0.f;
  for (int d = 0; d < 128; ++d) acc += bk[h * 128 + d] * q[d];
  c0[idx] = acc;
}

// s[i,h] = enc16[i,:] . u[g,h,:] + c0[g,h]   (masked -> -1e30)
__global__ __launch_bounds__(128)
void attn_s_kernel(const _Float16* __restrict__ enc, const int* __restrict__ keep,
                   const int* __restrict__ nodeb, const float* __restrict__ u,
                   const float* __restrict__ c0, float* __restrict__ sbuf) {
  int i = blockIdx.x;
  int h = threadIdx.x >> 5, lane = threadIdx.x & 31;
  if (!keep[i]) { if (lane == 0) sbuf[(size_t)i * 4 + h] = -1e30f; return; }
  int g = nodeb[i];
  const _Float16* er = enc + (size_t)i * D_;
  const float* ur = u + ((size_t)g * 4 + h) * D_;
  float acc = 0.f;
  for (int c = lane; c < D_; c += 32) acc += (float)er[c] * ur[c];
  for (int o = 16; o; o >>= 1) acc += __shfl_xor(acc, o, 32);
  if (lane == 0) sbuf[(size_t)i * 4 + h] = acc + c0[g * 4 + h];
}

// per-graph segment softmax: sbuf -> e (in place), den[g,h]
__global__ void attn_softmax_kernel(float* __restrict__ sbuf, const int* __restrict__ keep,
                                    float* __restrict__ den) {
  __shared__ float red[NPG_];
  int g = blockIdx.x, t = threadIdx.x;
  int i = g * NPG_ + t;
  int kp = keep[i];
  float sv[4], ev[4];
#pragma unroll
  for (int h = 0; h < 4; ++h) sv[h] = sbuf[(size_t)i * 4 + h];
#pragma unroll
  for (int h = 0; h < 4; ++h) {
    red[t] = sv[h];
    __syncthreads();
    for (int off = NPG_ / 2; off; off >>= 1) {
      if (t < off) red[t] = fmaxf(red[t], red[t + off]);
      __syncthreads();
    }
    float m = red[0];
    __syncthreads();
    float e = kp ? __expf(sv[h] - m) : 0.f;
    red[t] = e;
    __syncthreads();
    for (int off = NPG_ / 2; off; off >>= 1) {
      if (t < off) red[t] += red[t + off];
      __syncthreads();
    }
    if (t == 0) den[g * 4 + h] = red[0];
    __syncthreads();
    ev[h] = e;
  }
#pragma unroll
  for (int h = 0; h < 4; ++h) sbuf[(size_t)i * 4 + h] = ev[h];
}

// Z[g,h,c] = sum_{i in g} e[i,h] * enc16[i,c]
__global__ void attn_z_kernel(const float* __restrict__ ebuf, const _Float16* __restrict__ enc,
                              float* __restrict__ Z) {
  __shared__ float se[NPG_ * 4];
  int g = blockIdx.x, t = threadIdx.x; // 512 threads = c
  for (int j = t; j < NPG_ * 4; j += NPG_) se[j] = ebuf[(size_t)g * NPG_ * 4 + j];
  __syncthreads();
  float a0 = 0.f, a1 = 0.f, a2 = 0.f, a3 = 0.f;
  const _Float16* ep = enc + (size_t)g * NPG_ * D_ + t;
  for (int j = 0; j < NPG_; ++j) {
    float x = (float)ep[(size_t)j * D_];
    a0 += se[j * 4 + 0] * x;
    a1 += se[j * 4 + 1] * x;
    a2 += se[j * 4 + 2] * x;
    a3 += se[j * 4 + 3] * x;
  }
  Z[((size_t)g * 4 + 0) * D_ + t] = a0;
  Z[((size_t)g * 4 + 1) * D_ + t] = a1;
  Z[((size_t)g * 4 + 2) * D_ + t] = a2;
  Z[((size_t)g * 4 + 3) * D_ + t] = a3;
}

// o[g, h*128+d] = (Z[g,h,:].wv[h*128+d,:]) / den[g,h] + bv[h*128+d]
__global__ void attn_o_kernel(const float* __restrict__ Z, const float* __restrict__ den,
                              const float* __restrict__ wv, const float* __restrict__ bv,
                              float* __restrict__ o) {
  int idx = blockIdx.x * blockDim.x + threadIdx.x; // G*512
  if (idx >= G_ * D_) return;
  int g = idx >> 9, n = idx & 511, h = n >> 7;
  const float* z = Z + ((size_t)g * 4 + h) * D_;
  const float* w = wv + (size_t)n * D_;
  float acc = 0.f;
  for (int c = 0; c < D_; ++c) acc += z[c] * w[c];
  o[idx] = acc / den[g * 4 + h] + bv[n];
}

// ---------------- comment_attn (small) ----------------

__global__ void s2_kernel(const float* __restrict__ kh2, const float* __restrict__ qh2,
                          const int* __restrict__ cbatch, float* __restrict__ s2) {
  int idx = blockIdx.x * blockDim.x + threadIdx.x; // G*4
  if (idx >= G_ * 4) return;
  int h = idx & 3, g = idx >> 2;
  int b = cbatch[g];
  const float* k = kh2 + (size_t)g * D_ + h * 128;
  const float* q = qh2 + (size_t)b * D_ + h * 128;
  float acc = 0.f;
  for (int d = 0; d < 128; ++d) acc += k[d] * q[d];
  s2[idx] = acc;
}

__global__ void softmax2_kernel(float* __restrict__ s2, const int* __restrict__ cbatch,
                                float* __restrict__ den2) {
  int idx = blockIdx.x * blockDim.x + threadIdx.x; // B*4
  if (idx >= B_ * 4) return;
  int h = idx & 3, b = idx >> 2;
  float m = -1e30f;
  for (int g = 0; g < G_; ++g)
    if (cbatch[g] == b) m = fmaxf(m, s2[g * 4 + h]);
  float dn = 0.f;
  for (int g = 0; g < G_; ++g)
    if (cbatch[g] == b) {
      float e = __expf(s2[g * 4 + h] - m);
      s2[g * 4 + h] = e;
      dn += e;
    }
  den2[idx] = dn;
}

__global__ void o2_kernel(const float* __restrict__ e2, const float* __restrict__ den2,
                          const float* __restrict__ vh2, const int* __restrict__ cbatch,
                          float* __restrict__ o2) {
  int idx = blockIdx.x * blockDim.x + threadIdx.x; // B*512
  if (idx >= B_ * D_) return;
  int b = idx / D_, n = idx % D_, h = n >> 7;
  float acc = 0.f;
  for (int g = 0; g < G_; ++g)
    if (cbatch[g] == b) acc += e2[g * 4 + h] * vh2[(size_t)g * D_ + n];
  o2[idx] = acc / den2[b * 4 + h];
}

__global__ void hidden_kernel(const float* __restrict__ news, const float* __restrict__ fin,
                              const float* __restrict__ w1, const float* __restrict__ b1,
                              float* __restrict__ hid) {
  int idx = blockIdx.x * blockDim.x + threadIdx.x; // B*512
  if (idx >= B_ * D_) return;
  int r = idx / D_, n = idx % D_;
  const float* w = w1 + (size_t)n * (2 * D_);
  float acc = b1[n];
  for (int m = 0; m < D_; ++m) acc += news[(size_t)r * D_ + m] * w[m];
  for (int m = 0; m < D_; ++m) acc += fin[(size_t)r * D_ + m] * w[D_ + m];
  hid[idx] = fmaxf(acc, 0.f);
}

__global__ void loss_kernel(const float* __restrict__ pred, const int* __restrict__ label,
                            float* __restrict__ out) {
  __shared__ float red[B_];
  int b = threadIdx.x;
  float p0 = pred[b * 2 + 0], p1 = pred[b * 2 + 1];
  out[b * 2 + 0] = p0;
  out[b * 2 + 1] = p1;
  float m = fmaxf(p0, p1);
  float lse = m + logf(__expf(p0 - m) + __expf(p1 - m));
  float lp = (label[b] ? p1 : p0) - lse;
  red[b] = lp;
  __syncthreads();
  if (b == 0) {
    float s = 0.f;
    for (int i = 0; i < B_; ++i) s += red[i];
    out[B_ * 2] = -s / (float)B_;
  }
}

// ---------------- host driver ----------------

extern "C" void kernel_launch(void* const* d_in, const int* in_sizes, int n_in,
                              void* d_out, int out_size, void* d_ws, size_t ws_size,
                              hipStream_t stream) {
  (void)in_sizes; (void)n_in; (void)out_size; (void)ws_size;
  const float* emb       = (const float*)d_in[0];
  const int*   cx        = (const int*)d_in[1];
  const int*   cedge     = (const int*)d_in[2];
  const int*   mx        = (const int*)d_in[4];
  const int*   medge     = (const int*)d_in[5];
  const int*   mnodeb    = (const int*)d_in[6];
  const int*   mbatch    = (const int*)d_in[7];
  const int*   label     = (const int*)d_in[8];
  const float* wc1 = (const float*)d_in[9],  *bc1 = (const float*)d_in[10];
  const float* wc2 = (const float*)d_in[11], *bc2 = (const float*)d_in[12];
  const float* wtr = (const float*)d_in[13], *btr = (const float*)d_in[14];
  const float* w_score = (const float*)d_in[15], *b_score = (const float*)d_in[16];
  const float* wenc = (const float*)d_in[17], *benc = (const float*)d_in[18];
  const float* na_wq = (const float*)d_in[19], *na_bq = (const float*)d_in[20];
  const float* na_wk = (const float*)d_in[21], *na_bk = (const float*)d_in[22];
  const float* na_wv = (const float*)d_in[23], *na_bv = (const float*)d_in[24];
  const float* na_wo = (const float*)d_in[25], *na_bo = (const float*)d_in[26];
  const float* ca_wq = (const float*)d_in[27], *ca_bq = (const float*)d_in[28];
  const float* ca_wk = (const float*)d_in[29], *ca_bk = (const float*)d_in[30];
  const float* ca_wv = (const float*)d_in[31], *ca_bv = (const float*)d_in[32];
  const float* ca_wo = (const float*)d_in[33], *ca_bo = (const float*)d_in[34];
  const float* w1 = (const float*)d_in[35], *b1 = (const float*)d_in[36];
  const float* w2 = (const float*)d_in[37], *b2 = (const float*)d_in[38];
  float* out = (float*)d_out;

  char* w = (char*)d_ws;
  size_t off = 0;
  auto alloc = [&](size_t bytes) -> void* {
    void* p = w + off;
    off += (bytes + 255) & ~(size_t)255;
    return p;
  };
  float*    HBUF  = (float*)alloc((size_t)NM_ * D_ * 4);
  float*    AGG   = (float*)alloc((size_t)NM_ * D_ * 4);
  _Float16* X16A  = (_Float16*)alloc((size_t)NM_ * KP_ * 2);
  _Float16* X16B  = (_Float16*)alloc((size_t)NM_ * D_ * 2);
  _Float16* wc1h  = (_Float16*)alloc((size_t)D_ * KP_ * 2);
  _Float16* wc2h  = (_Float16*)alloc((size_t)D_ * D_ * 2);
  _Float16* wench = (_Float16*)alloc((size_t)D_ * KP_ * 2);
  float* degC = (float*)alloc(NC_ * 4);
  float* disC = (float*)alloc(NC_ * 4);
  float* degM = (float*)alloc(NM_ * 4);
  float* disM = (float*)alloc(NM_ * 4);
  float* degE = (float*)alloc(NM_ * 4);
  float* disE = (float*)alloc(NM_ * 4);
  float* tb   = (float*)alloc(NM_ * 4);
  float* scoreb = (float*)alloc(NM_ * 4);
  int*   keep = (int*)alloc(NM_ * 4);
  float* wsv  = (float*)alloc(304 * 4);
  float* news = (float*)alloc(B_ * D_ * 4);
  float* qh   = (float*)alloc(G_ * D_ * 4);
  float* ub   = (float*)alloc((size_t)G_ * 4 * D_ * 4);
  float* c0   = (float*)alloc(G_ * 4 * 4);
  float* sb   = (float*)alloc((size_t)NM_ * 4 * 4);
  float* den  = (float*)alloc(G_ * 4 * 4);
  float* Zb   = (float*)alloc((size_t)G_ * 4 * D_ * 4);
  float* ob   = (float*)alloc(G_ * D_ * 4);
  float* gout = (float*)alloc(G_ * D_ * 4);
  float* qh2  = (float*)alloc(B_ * D_ * 4);
  float* kh2  = (float*)alloc(G_ * D_ * 4);
  float* vh2  = (float*)alloc(G_ * D_ * 4);
  float* s2   = (float*)alloc(G_ * 4 * 4);
  float* den2 = (float*)alloc(B_ * 4 * 4);
  float* o2   = (float*)alloc(B_ * D_ * 4);
  float* fin  = (float*)alloc(B_ * D_ * 4);
  float* hid  = (float*)alloc(B_ * D_ * 4);
  float* predb = (float*)alloc(B_ * 2 * 4);

  // weight conversions
  wconv_kernel<<<CDIV(D_ * KP_, 256), 256, 0, stream>>>(wc1, wc1h, D_, KE_, KP_);
  wconv_kernel<<<CDIV(D_ * D_, 256), 256, 0, stream>>>(wc2, wc2h, D_, D_, D_);
  wconv_kernel<<<CDIV(D_ * KP_, 256), 256, 0, stream>>>(wenc, wench, D_, KE_, KP_);

  // ---- content branch ----
  {
    long long tot = (long long)NC_ * KP_;
    gather_half_kernel<<<CDIV(tot, 256), 256, 0, stream>>>(emb, cx, X16A, tot, KP_);
  }
  gemm_f16_wmma<<<dim3(NC_ / 64, 8), 128, 0, stream>>>(X16A, wc1h, HBUF, KP_);
  zero_kernel<<<CDIV(NC_, 256), 256, 0, stream>>>(degC, NC_);
  edge_deg_unit_kernel<<<CDIV(EC_, 256), 256, 0, stream>>>(cedge, degC, EC_);
  degfin_kernel<<<CDIV(NC_, 256), 256, 0, stream>>>(degC, disC, NC_);
  init_gcn_kernel<<<CDIV((long long)NC_ * D_, 256), 256, 0, stream>>>(AGG, HBUF, degC, bc1, (long long)NC_ * D_);
  edge_agg_unit_kernel<<<EC_, 256, 0, stream>>>(cedge, HBUF, disC, AGG, EC_);
  relu_half_kernel<<<CDIV((long long)NC_ * D_, 256), 256, 0, stream>>>(AGG, X16B, (long long)NC_ * D_);
  gemm_f16_wmma<<<dim3(NC_ / 64, 8), 128, 0, stream>>>(X16B, wc2h, HBUF, D_);
  init_gcn_kernel<<<CDIV((long long)NC_ * D_, 256), 256, 0, stream>>>(AGG, HBUF, degC, bc2, (long long)NC_ * D_);
  edge_agg_unit_kernel<<<EC_, 256, 0, stream>>>(cedge, HBUF, disC, AGG, EC_);
  meanpool_kernel<<<dim3(B_, 8), 64, 0, stream>>>(AGG, news);

  // ---- comments: score path + pooling ----
  {
    long long tot = (long long)NM_ * KP_;
    gather_half_kernel<<<CDIV(tot, 256), 256, 0, stream>>>(emb, mx, X16A, tot, KP_);
  }
  zero_kernel<<<CDIV(NM_, 256), 256, 0, stream>>>(degM, NM_);
  edge_deg_unit_kernel<<<CDIV(EM_, 256), 256, 0, stream>>>(medge, degM, EM_);
  degfin_kernel<<<CDIV(NM_, 256), 256, 0, stream>>>(degM, disM, NM_);
  scorevec_kernel<<<2, 256, 0, stream>>>(wtr, btr, w_score, b_score, wsv);
  tvec_kernel<<<CDIV(NM_, 256), 256, 0, stream>>>(emb, mx, wsv, tb);
  score_init_kernel<<<CDIV(NM_, 256), 256, 0, stream>>>(tb, degM, wsv, scoreb);
  edge_score_kernel<<<CDIV(EM_, 256), 256, 0, stream>>>(medge, tb, disM, scoreb, EM_);
  keep_kernel<<<G_, NPG_, 0, stream>>>(scoreb, keep);

  // ---- comments: enc GCN on pooled subgraph ----
  zero_kernel<<<CDIV(NM_, 256), 256, 0, stream>>>(degE, NM_);
  edge_deg_mask_kernel<<<CDIV(EM_, 256), 256, 0, stream>>>(medge, keep, degE, EM_);
  degfin_kernel<<<CDIV(NM_, 256), 256, 0, stream>>>(degE, disE, NM_);
  gemm_f16_wmma<<<dim3(NM_ / 64, 8), 128, 0, stream>>>(X16A, wench, HBUF, KP_);
  init_gcn_kernel<<<CDIV((long long)NM_ * D_, 256), 256, 0, stream>>>(AGG, HBUF, degE, benc, (long long)NM_ * D_);
  edge_agg_mask_kernel<<<EM_, 256, 0, stream>>>(medge, keep, HBUF, disE, AGG, EM_);
  relu_half_kernel<<<CDIV((long long)NM_ * D_, 256), 256, 0, stream>>>(AGG, X16B, (long long)NM_ * D_);

  // ---- node attention ----
  linear_kernel<<<CDIV(G_ * D_, 256), 256, 0, stream>>>(news, mbatch, na_wq, na_bq, qh, G_, D_, D_, QSCALE_, 0);
  attn_u_kernel<<<CDIV(G_ * 4 * D_, 256), 256, 0, stream>>>(qh, na_wk, ub);
  attn_c0_kernel<<<CDIV(G_ * 4, 256), 256, 0, stream>>>(qh, na_bk, c0);
  attn_s_kernel<<<NM_, 128, 0, stream>>>(X16B, keep, mnodeb, ub, c0, sb);
  attn_softmax_kernel<<<G_, NPG_, 0, stream>>>(sb, keep, den);
  attn_z_kernel<<<G_, D_, 0, stream>>>(sb, X16B, Zb);
  attn_o_kernel<<<CDIV(G_ * D_, 256), 256, 0, stream>>>(Zb, den, na_wv, na_bv, ob);
  linear_kernel<<<CDIV(G_ * D_, 256), 256, 0, stream>>>(ob, nullptr, na_wo, na_bo, gout, G_, D_, D_, 1.f, 0);

  // ---- comment attention ----
  linear_kernel<<<CDIV(B_ * D_, 256), 256, 0, stream>>>(news, nullptr, ca_wq, ca_bq, qh2, B_, D_, D_, QSCALE_, 0);
  linear_kernel<<<CDIV(G_ * D_, 256), 256, 0, stream>>>(gout, nullptr, ca_wk, ca_bk, kh2, G_, D_, D_, 1.f, 0);
  linear_kernel<<<CDIV(G_ * D_, 256), 256, 0, stream>>>(gout, nullptr, ca_wv, ca_bv, vh2, G_, D_, D_, 1.f, 0);
  s2_kernel<<<CDIV(G_ * 4, 256), 256, 0, stream>>>(kh2, qh2, mbatch, s2);
  softmax2_kernel<<<1, B_ * 4, 0, stream>>>(s2, mbatch, den2);
  o2_kernel<<<CDIV(B_ * D_, 256), 256, 0, stream>>>(s2, den2, vh2, mbatch, o2);
  linear_kernel<<<CDIV(B_ * D_, 256), 256, 0, stream>>>(o2, nullptr, ca_wo, ca_bo, fin, B_, D_, D_, 1.f, 0);

  // ---- classifier + loss ----
  hidden_kernel<<<CDIV(B_ * D_, 256), 256, 0, stream>>>(news, fin, w1, b1, hid);
  linear_kernel<<<1, 64, 0, stream>>>(hid, nullptr, w2, b2, predb, B_, 2, D_, 1.f, 0);
  loss_kernel<<<1, B_, 0, stream>>>(predb, label, out);
}
